// Decoder_2860448219142
// MI455X (gfx1250) — matrix operand order repaired and checked by reading, not compiled, not run on hardware
//
#include <hip/hip_runtime.h>

// ---------------------------------------------------------------------------
// Performer (FAVOR+) decoder layer for MI455X (gfx1250, wave32, WMMA).
// All dense math runs on v_wmma_f32_16x16x32_bf16 with f32 accumulation.
// A-operand tiles are staged global->LDS by the Tensor Data Mover
// (tensor_load_to_lds + s_wait_tensorcnt); W tiles are transposed manually.
// Deterministic: no floating-point atomics; K-split partials are reduced by a
// fixed-order reduction kernel. Workspace requirement ~742 MB.
// ---------------------------------------------------------------------------

typedef unsigned short u16;
typedef __attribute__((ext_vector_type(16))) __bf16 v16bf;
typedef __attribute__((ext_vector_type(8)))  __bf16 v8bf;
typedef __attribute__((ext_vector_type(8)))  float  v8f;
typedef __attribute__((ext_vector_type(4)))  unsigned su4;
typedef __attribute__((ext_vector_type(8)))  unsigned su8;

constexpr int kBat = 4, kN = 4096, kH = 1024, kNH = 16, kHD = 64, kM = 256;
constexpr int kBH = kBat * kNH;                       // 64 (batch*heads)
constexpr size_t kACT = (size_t)kBat * kN * kH;       // 16,777,216
constexpr size_t kQP  = (size_t)kBH * kN * kM;        // 67,108,864 (== 4*kACT)
constexpr size_t kCTX = (size_t)kBH * kM * kHD;       // 1,048,576
constexpr int kKSPLIT = 8;

// ---------------- small device helpers -------------------------------------
__device__ __forceinline__ u16 f2bf(float f) {
  unsigned u = __float_as_uint(f);
  u += 0x7FFFu + ((u >> 16) & 1u);          // round-to-nearest-even
  return (u16)(u >> 16);
}
__device__ __forceinline__ float bf2f(u16 h) {
  return __uint_as_float((unsigned)h << 16);
}
// Build a 16-element bf16 fragment from two 16B LDS loads.
__device__ __forceinline__ v16bf frag16(const u16* p0, const u16* p1) {
  v8bf lo = *(const v8bf*)p0;
  v8bf hi = *(const v8bf*)p1;
  return __builtin_shufflevector(lo, hi, 0, 1, 2, 3, 4, 5, 6, 7,
                                 8, 9, 10, 11, 12, 13, 14, 15);
}
__device__ __forceinline__ v8f wmma_bf16(v16bf a, v16bf b, v8f c) {
  return __builtin_amdgcn_wmma_f32_16x16x32_bf16(false, a, false, b,
                                                 (short)0, c, false, false);
}
__device__ __forceinline__ float gelu_tanh(float x) {
  float t = 0.7978845608028654f * (x + 0.044715f * x * x * x);
  return 0.5f * x * (1.0f + tanhf(t));
}

// Tensor Data Mover: issue an async 2D bf16 tile load (tile_w elems per row,
// tile_h rows, row stride stride_elems) into LDS with hardware row padding
// (64B data + 16B pad per row -> the 40-halfword padded LDS rows used below).
// D# group0/group1 per CDNA5 ISA; groups 2/3 omitted (2D).
__device__ __forceinline__ void tdm_issue_2d(const u16* gsrc, unsigned lds_addr,
                                             unsigned tile_w, unsigned tile_h,
                                             unsigned stride_elems) {
  unsigned long long ga = (unsigned long long)(size_t)gsrc;
  su4 g0;
  g0[0] = 1u;                                  // count=1 valid user descriptor
  g0[1] = lds_addr;                            // LDS byte address
  g0[2] = (unsigned)ga;                        // global_addr[31:0]
  g0[3] = (unsigned)(ga >> 32) | (2u << 30);   // global_addr[56:32], type=2
  su8 g1;
  // data_size=1 (2B), pad_enable=1, pad_interval=3 (16 DW), pad_amount=3 (4 DW)
  g1[0] = (1u << 16) | (1u << 20) | (3u << 22) | (3u << 25);
  g1[1] = 0u;                          // tensor_dim0[15:0]=0 (dim=0x40000000)
  g1[2] = 0x4000u;                     // tensor_dim0[31:16]; tensor_dim1 lo=0
  g1[3] = 0x4000u | (tile_w << 16);    // tensor_dim1[31:16]; tile_dim0
  g1[4] = tile_h;                      // tile_dim1; tile_dim2=0
  g1[5] = stride_elems;                // tensor_dim0_stride[31:0]
  g1[6] = 0u;                          // stride0 hi, tensor_dim1_stride lo
  g1[7] = 0u;
  asm volatile("tensor_load_to_lds %0, %1" ::"s"(g0), "s"(g1) : "memory");
}

// ---------------- generic bf16 WMMA GEMM ------------------------------------
// C[Mr x Nc] = A[Mr x K] * W[K x Nc] + bias[Nc]
// EPI: 0 = bias -> bf16 out, 1 = bias + resid -> f32 out, 2 = bias+GELU -> bf16
// Block tile 128x128, BK=32, 8 waves, each wave owns a 32x64 sub-tile (2x4 WMMA).
template <int EPI>
__global__ __launch_bounds__(256) void gemm_bf16_kernel(
    const u16* __restrict__ A, const u16* __restrict__ W,
    const float* __restrict__ bias, const float* __restrict__ resid,
    float* __restrict__ Cf, u16* __restrict__ Cbf, int Mr, int Nc, int K) {
  __shared__ u16 As[128 * 40];   // [row][k]  (padded to 40 halfwords)
  __shared__ u16 Bs[128 * 40];   // [col][k]  (W tile stored transposed)
  const int tid = threadIdx.x, lane = tid & 31, wave = tid >> 5;
  const int lr = lane & 15, lh = lane >> 4;
  const int m0 = blockIdx.y * 128, n0 = blockIdx.x * 128;
  const int wm0 = (wave & 3) * 32, wn0 = (wave >> 2) * 64;
  const unsigned asBase = (unsigned)(size_t)&As[0];

  v8f acc[2][4];
  for (int i = 0; i < 2; ++i)
    for (int j = 0; j < 4; ++j)
      for (int e = 0; e < 8; ++e) acc[i][j][e] = 0.0f;

  for (int k0 = 0; k0 < K; k0 += 32) {
    // A tile: 128x32 bf16 via Tensor Data Mover (one issue per block)
    if (wave == 0)
      tdm_issue_2d(A + (size_t)m0 * K + k0, asBase, 32u, 128u, (unsigned)K);
    // W tile: 32x128 bf16, stored transposed into Bs[col][k]
    for (int c = tid; c < 512; c += 256) {
      int kk = c >> 4, co = (c & 15) << 3;
      uint4 v = *(const uint4*)(W + (size_t)(k0 + kk) * Nc + n0 + co);
      if (k0 + 32 < K)  // prefetch next K-step of the W stream
        __builtin_prefetch(W + (size_t)(k0 + 32 + kk) * Nc + n0 + co, 0, 0);
      const u16* pv = (const u16*)&v;
#pragma unroll
      for (int i = 0; i < 8; ++i) Bs[(co + i) * 40 + kk] = pv[i];
    }
    if (wave == 0) __builtin_amdgcn_s_wait_tensorcnt(0);
    __syncthreads();
    v16bf af[2], bfv[4];
#pragma unroll
    for (int i = 0; i < 2; ++i) {
      const u16* rb = &As[(wm0 + i * 16 + lr) * 40];
      af[i] = frag16(rb + lh * 16, rb + lh * 16 + 8);
    }
#pragma unroll
    for (int j = 0; j < 4; ++j) {
      const u16* cb = &Bs[(wn0 + j * 16 + lr) * 40 + lh * 16];
      bfv[j] = frag16(cb, cb + 8);
    }
#pragma unroll
    for (int i = 0; i < 2; ++i)
#pragma unroll
      for (int j = 0; j < 4; ++j)
        acc[i][j] = wmma_bf16(af[i], bfv[j], acc[i][j]);
    __syncthreads();
  }
#pragma unroll
  for (int i = 0; i < 2; ++i) {
#pragma unroll
    for (int j = 0; j < 4; ++j) {
      const int col = n0 + wn0 + j * 16 + lr;
      const float bv = bias[col];
#pragma unroll
      for (int e = 0; e < 8; ++e) {
        const int row = m0 + wm0 + i * 16 + lh * 8 + e;
        const size_t idx = (size_t)row * Nc + col;
        float v = acc[i][j][e] + bv;
        if (EPI == 0) Cbf[idx] = f2bf(v);
        if (EPI == 1) Cf[idx] = v + resid[idx];
        if (EPI == 2) Cbf[idx] = f2bf(gelu_tanh(v));
      }
    }
  }
}

// ---------------- FAVOR+ feature map ----------------------------------------
// dd[n,m] = (data[n,:]*dn) . proj[m,:];  P = M^-1/2 * (exp(dd - diag - rowmax)+1e-4)
// Block = one (b,h) and 64 rows of n; all 256 feature columns resident.
// TRANS=0: store P as Qp[bh][n][m].  TRANS=1: store P as KpT[bh][m][n].
template <int TRANS>
__global__ __launch_bounds__(256) void feat_kernel(
    const u16* __restrict__ data, const float* __restrict__ proj,
    u16* __restrict__ P) {
  __shared__ u16 As[64 * 72];       // (Q*dn) bf16 [row][d], padded
  __shared__ u16 Bs[256 * 72];      // proj bf16 [m][d]  (already B-transposed)
  __shared__ float diag_s[64];
  __shared__ float pm[64 * 2];      // per-128-col row-max partials
  const int tid = threadIdx.x, lane = tid & 31, wave = tid >> 5;
  const int lr = lane & 15, lh = lane >> 4;
  const int bh = blockIdx.x, n0 = blockIdx.y * 64;
  const int b = bh / kNH, h = bh % kNH;
  const float dn = 0.3535533905932738f;   // 64^-0.25
  const float rsM = 0.0625f;              // 256^-0.5

  for (int c = tid; c < 512; c += 256) {  // 64x64 elems
    int r = c >> 3, ko = (c & 7) << 3;
    uint4 v = *(const uint4*)(data + ((size_t)(b * kN + n0 + r)) * kH +
                              h * kHD + ko);
    const u16* pv = (const u16*)&v;
    union { u16 t[8]; uint4 q; } u;
#pragma unroll
    for (int i = 0; i < 8; ++i) u.t[i] = f2bf(bf2f(pv[i]) * dn);
    *(uint4*)&As[r * 72 + ko] = u.q;
  }
  if (tid < 64) {
    const u16* g = data + ((size_t)(b * kN + n0 + tid)) * kH + h * kHD;
    float s = 0.f;
    for (int d = 0; d < kHD; ++d) { float v = bf2f(g[d]); s += v * v; }
    diag_s[tid] = 0.5f * dn * dn * s;
  }
  for (int c = tid; c < 2048; c += 256) {  // 256x64 proj elems
    int m = c >> 3, ko = (c & 7) << 3;
    const float* g = proj + m * kHD + ko;
    union { u16 t[8]; uint4 q; } u;
#pragma unroll
    for (int i = 0; i < 8; ++i) u.t[i] = f2bf(g[i]);
    *(uint4*)&Bs[m * 72 + ko] = u.q;
  }
  __syncthreads();

  const int rt = wave >> 1;            // row tile 0..3
  const int cb = (wave & 1) * 128;     // column base
  v8f acc[8];
  for (int j = 0; j < 8; ++j)
    for (int e = 0; e < 8; ++e) acc[j][e] = 0.0f;
#pragma unroll
  for (int kk = 0; kk < 2; ++kk) {
    const int k0 = kk * 32;
    const u16* rb = &As[(rt * 16 + lr) * 72 + k0];
    v16bf af = frag16(rb + lh * 16, rb + lh * 16 + 8);
#pragma unroll
    for (int j = 0; j < 8; ++j) {
      const u16* cp = &Bs[(cb + j * 16 + lr) * 72 + k0 + lh * 16];
      acc[j] = wmma_bf16(af, frag16(cp, cp + 8), acc[j]);
    }
  }
  // row-max: in-register across col tiles, then cross-lane (16 cols), then LDS
  float rpart[8];
#pragma unroll
  for (int e = 0; e < 8; ++e) {
    float m = acc[0][e];
#pragma unroll
    for (int j = 1; j < 8; ++j) m = fmaxf(m, acc[j][e]);
#pragma unroll
    for (int msk = 1; msk <= 8; msk <<= 1) m = fmaxf(m, __shfl_xor(m, msk, 32));
    rpart[e] = m;
  }
  if (lr == 0) {
#pragma unroll
    for (int e = 0; e < 8; ++e)
      pm[(rt * 16 + lh * 8 + e) * 2 + (wave & 1)] = rpart[e];
  }
  __syncthreads();
#pragma unroll
  for (int j = 0; j < 8; ++j) {
    const int col = cb + j * 16 + lr;
#pragma unroll
    for (int e = 0; e < 8; ++e) {
      const int row = rt * 16 + lh * 8 + e;
      const float rmax = fmaxf(pm[row * 2], pm[row * 2 + 1]);
      const float v =
          rsM * (__expf(acc[j][e] - diag_s[row] - rmax) + 1e-4f);
      if (TRANS == 0)
        P[((size_t)bh * kN + n0 + row) * kM + col] = f2bf(v);
      else
        P[((size_t)bh * kM + col) * kN + n0 + row] = f2bf(v);
    }
  }
}

// Ksum[bh][m] = sum_n KpT[bh][m][n]  (deterministic block reduction)
__global__ __launch_bounds__(256) void ksum_kernel(const u16* __restrict__ KpT,
                                                   float* __restrict__ Ksum) {
  __shared__ float red[256];
  const int tid = threadIdx.x, row = blockIdx.x;  // row = bh*256+m
  const u16* p = KpT + (size_t)row * kN;
  float s = 0.f;
  for (int i = tid; i < kN; i += 256) s += bf2f(p[i]);
  red[tid] = s;
  __syncthreads();
  for (int w = 128; w > 0; w >>= 1) {
    if (tid < w) red[tid] += red[tid + w];
    __syncthreads();
  }
  if (tid == 0) Ksum[row] = red[0];
}

// ctx partials: ctxp[ks][bh][m][d] = KpT[bh][256 x Nslice] * V[Nslice x 64]
__global__ __launch_bounds__(256) void ctx_kernel(const u16* __restrict__ KpT,
                                                  const u16* __restrict__ vbuf,
                                                  float* __restrict__ ctxp) {
  __shared__ u16 As[256 * 40];   // KpT tile [m][k]
  __shared__ u16 Bs[64 * 40];    // V^T tile [d][k]
  const int tid = threadIdx.x, lane = tid & 31, wave = tid >> 5;
  const int lr = lane & 15, lh = lane >> 4;
  const int bh = blockIdx.x, ks = blockIdx.y;
  const int b = bh / kNH, h = bh % kNH;
  const int wm0 = wave * 32;
  const unsigned asBase = (unsigned)(size_t)&As[0];
  v8f acc[2][4];
  for (int i = 0; i < 2; ++i)
    for (int j = 0; j < 4; ++j)
      for (int e = 0; e < 8; ++e) acc[i][j][e] = 0.0f;

  const int kbeg = ks * (kN / kKSPLIT), kend = kbeg + kN / kKSPLIT;
  for (int k0 = kbeg; k0 < kend; k0 += 32) {
    // KpT tile 256x32 via TDM
    if (wave == 0)
      tdm_issue_2d(KpT + (size_t)bh * kM * kN + k0, asBase, 32u, 256u,
                   (unsigned)kN);
    {  // V rows n=k0..k0+31, 64 cols -> transposed into Bs
      int kk = tid >> 3, dof = (tid & 7) << 3;
      uint4 v = *(const uint4*)(vbuf + ((size_t)(b * kN + k0 + kk)) * kH +
                                h * kHD + dof);
      const u16* pv = (const u16*)&v;
#pragma unroll
      for (int i = 0; i < 8; ++i) Bs[(dof + i) * 40 + kk] = pv[i];
    }
    if (wave == 0) __builtin_amdgcn_s_wait_tensorcnt(0);
    __syncthreads();
    v16bf af[2], bfv[4];
#pragma unroll
    for (int i = 0; i < 2; ++i) {
      const u16* rb = &As[(wm0 + i * 16 + lr) * 40];
      af[i] = frag16(rb + lh * 16, rb + lh * 16 + 8);
    }
#pragma unroll
    for (int j = 0; j < 4; ++j) {
      const u16* cp = &Bs[(j * 16 + lr) * 40 + lh * 16];
      bfv[j] = frag16(cp, cp + 8);
    }
#pragma unroll
    for (int i = 0; i < 2; ++i)
#pragma unroll
      for (int j = 0; j < 4; ++j)
        acc[i][j] = wmma_bf16(af[i], bfv[j], acc[i][j]);
    __syncthreads();
  }
  float* outp = ctxp + (size_t)ks * kCTX;
#pragma unroll
  for (int i = 0; i < 2; ++i)
#pragma unroll
    for (int j = 0; j < 4; ++j)
#pragma unroll
      for (int e = 0; e < 8; ++e) {
        const int m = wm0 + i * 16 + lh * 8 + e, d = j * 16 + lr;
        outp[((size_t)bh * kM + m) * kHD + d] = acc[i][j][e];
      }
}

__global__ void ctx_reduce_kernel(const float* __restrict__ ctxp,
                                  float* __restrict__ ctx) {
  size_t i = (size_t)blockIdx.x * 256 + threadIdx.x;
  float s = 0.f;
#pragma unroll
  for (int p = 0; p < kKSPLIT; ++p) s += ctxp[(size_t)p * kCTX + i];
  ctx[i] = s;
}

// out[bh][n][d] = (Qp[bh][n][:] . ctx[bh][:][d]) / (Qp[bh][n][:] . Ksum[bh][:])
__global__ __launch_bounds__(256) void attn_out_kernel(
    const u16* __restrict__ Qp, const float* __restrict__ ctx,
    const float* __restrict__ Ksum, u16* __restrict__ Obf) {
  __shared__ u16 As[128 * 40];
  __shared__ u16 Bs[64 * 40];
  __shared__ float ks_s[256];
  __shared__ float denp[256];
  __shared__ float den[128];
  const int tid = threadIdx.x, lane = tid & 31, wave = tid >> 5;
  const int lr = lane & 15, lh = lane >> 4;
  const int bh = blockIdx.x, n0 = blockIdx.y * 128;
  const unsigned asBase = (unsigned)(size_t)&As[0];
  ks_s[tid] = Ksum[bh * kM + tid];
  __syncthreads();
  {
    const int row = tid >> 1, hf = tid & 1;
    const u16* q = Qp + ((size_t)bh * kN + n0 + row) * kM + hf * 128;
    float s = 0.f;
    for (int m = 0; m < 128; ++m) s += bf2f(q[m]) * ks_s[hf * 128 + m];
    denp[row * 2 + hf] = s;
  }
  __syncthreads();
  if (tid < 128) den[tid] = denp[tid * 2] + denp[tid * 2 + 1];
  __syncthreads();

  v8f acc[4];
  for (int j = 0; j < 4; ++j)
    for (int e = 0; e < 8; ++e) acc[j][e] = 0.0f;
  for (int k0 = 0; k0 < kM; k0 += 32) {
    // Qp tile 128x32 via TDM
    if (wave == 0)
      tdm_issue_2d(Qp + ((size_t)bh * kN + n0) * kM + k0, asBase, 32u, 128u,
                   (unsigned)kM);
    {  // ctx rows m=k0..k0+31 x 64 d -> transposed bf16
      int kk = tid >> 3, dof = (tid & 7) << 3;
      const float* g = ctx + ((size_t)bh * kM + k0 + kk) * kHD + dof;
#pragma unroll
      for (int i = 0; i < 8; ++i) Bs[(dof + i) * 40 + kk] = f2bf(g[i]);
    }
    if (wave == 0) __builtin_amdgcn_s_wait_tensorcnt(0);
    __syncthreads();
    const u16* rb = &As[(wave * 16 + lr) * 40];
    v16bf af = frag16(rb + lh * 16, rb + lh * 16 + 8);
#pragma unroll
    for (int j = 0; j < 4; ++j) {
      const u16* cp = &Bs[(j * 16 + lr) * 40 + lh * 16];
      acc[j] = wmma_bf16(af, frag16(cp, cp + 8), acc[j]);
    }
    __syncthreads();
  }
#pragma unroll
  for (int j = 0; j < 4; ++j)
#pragma unroll
    for (int e = 0; e < 8; ++e) {
      const int row = wave * 16 + lh * 8 + e, d = j * 16 + lr;
      const float v = acc[j][e] / den[row];
      // [b][h][n][d] contiguous == reference's no-transpose reshape(b,-1,H)
      Obf[((size_t)bh * kN + n0 + row) * kHD + d] = f2bf(v);
    }
}

// LayerNorm over H=1024; writes f32 and (optionally) a bf16 copy.
__global__ __launch_bounds__(256) void ln_kernel(
    const float* __restrict__ X, const float* __restrict__ gam,
    const float* __restrict__ bet, float* __restrict__ Yf,
    u16* __restrict__ Ybf) {
  __shared__ float r1[256], r2[256];
  const int tid = threadIdx.x, row = blockIdx.x;
  const float* x = X + (size_t)row * kH;
  float s = 0.f, s2 = 0.f;
  for (int i = tid; i < kH; i += 256) {
    float v = x[i];
    s += v;
    s2 += v * v;
  }
  r1[tid] = s;
  r2[tid] = s2;
  __syncthreads();
  for (int w = 128; w > 0; w >>= 1) {
    if (tid < w) { r1[tid] += r1[tid + w]; r2[tid] += r2[tid + w]; }
    __syncthreads();
  }
  const float mu = r1[0] * (1.0f / kH);
  const float var = r2[0] * (1.0f / kH) - mu * mu;
  const float inv = rsqrtf(var + 1e-5f);
  for (int i = tid; i < kH; i += 256) {
    float v = (x[i] - mu) * inv * gam[i] + bet[i];
    Yf[(size_t)row * kH + i] = v;
    if (Ybf) Ybf[(size_t)row * kH + i] = f2bf(v);
  }
}

__global__ void cvt_bf16_kernel(const float* __restrict__ x,
                                u16* __restrict__ y, size_t n) {
  size_t i = (size_t)blockIdx.x * 256 + threadIdx.x;
  if (i < n) y[i] = f2bf(x[i]);
}

// ---------------------------------------------------------------------------
extern "C" void kernel_launch(void* const* d_in, const int* in_sizes, int n_in,
                              void* d_out, int out_size, void* d_ws,
                              size_t ws_size, hipStream_t stream) {
  const float* x       = (const float*)d_in[0];
  const float* enc     = (const float*)d_in[1];
  const float* sa_wq   = (const float*)d_in[2];
  const float* sa_bq   = (const float*)d_in[3];
  const float* sa_wk   = (const float*)d_in[4];
  const float* sa_bk   = (const float*)d_in[5];
  const float* sa_wv   = (const float*)d_in[6];
  const float* sa_bv   = (const float*)d_in[7];
  const float* sa_wo   = (const float*)d_in[8];
  const float* sa_bo   = (const float*)d_in[9];
  const float* sa_proj = (const float*)d_in[10];
  const float* ca_wq   = (const float*)d_in[11];
  const float* ca_bq   = (const float*)d_in[12];
  const float* ca_wk   = (const float*)d_in[13];
  const float* ca_bk   = (const float*)d_in[14];
  const float* ca_wv   = (const float*)d_in[15];
  const float* ca_bv   = (const float*)d_in[16];
  const float* ca_wo   = (const float*)d_in[17];
  const float* ca_bo   = (const float*)d_in[18];
  const float* ca_proj = (const float*)d_in[19];
  const float* ff_w1   = (const float*)d_in[20];
  const float* ff_b1   = (const float*)d_in[21];
  const float* ff_w2   = (const float*)d_in[22];
  const float* ff_b2   = (const float*)d_in[23];
  const float* ln1_g   = (const float*)d_in[24];
  const float* ln1_b   = (const float*)d_in[25];
  const float* ln2_g   = (const float*)d_in[26];
  const float* ln2_b   = (const float*)d_in[27];
  const float* ln3_g   = (const float*)d_in[28];
  const float* ln3_b   = (const float*)d_in[29];
  float* out = (float*)d_out;

  char* ws = (char*)d_ws;
  size_t off = 0;
  auto arena = [&](size_t bytes) -> void* {
    void* p = ws + off;
    off += (bytes + 255) & ~(size_t)255;
    return p;
  };
  const size_t WH = (size_t)kH * kH;
  u16* wq_s = (u16*)arena(WH * 2);
  u16* wk_s = (u16*)arena(WH * 2);
  u16* wv_s = (u16*)arena(WH * 2);
  u16* wo_s = (u16*)arena(WH * 2);
  u16* wq_c = (u16*)arena(WH * 2);
  u16* wk_c = (u16*)arena(WH * 2);
  u16* wv_c = (u16*)arena(WH * 2);
  u16* wo_c = (u16*)arena(WH * 2);
  u16* w1b  = (u16*)arena(WH * 4 * 2);
  u16* w2b  = (u16*)arena(WH * 4 * 2);
  u16* abfA = (u16*)arena(kACT * 2);   // current layer-input (bf16)
  u16* abfB = (u16*)arena(kACT * 2);   // attention output (bf16)
  u16* encb = (u16*)arena(kACT * 2);
  u16* qb   = (u16*)arena(kACT * 2);   // Q/K/V projections (bf16)
  u16* kb   = (u16*)arena(kACT * 2);
  u16* vb   = (u16*)arena(kACT * 2);
  float* t32 = (float*)arena(kACT * 4);  // pre-LN residual sums
  float* h1  = (float*)arena(kACT * 4);
  float* h2  = (float*)arena(kACT * 4);
  u16* Qp  = (u16*)arena(kQP * 2);
  u16* KpT = (u16*)arena(kQP * 2);
  float* Ksum = (float*)arena((size_t)kBH * kM * 4);
  float* ctxp = (float*)arena((size_t)kKSPLIT * kCTX * 4);
  float* ctxb = (float*)arena(kCTX * 4);
  u16* gelub = Qp;  // FF activations alias Qp (dead once attention is done)

  dim3 thr(256);
  const int Mr = kBat * kN;  // 16384 token rows

  auto cvt = [&](const float* s, u16* d, size_t n) {
    cvt_bf16_kernel<<<dim3((unsigned)((n + 255) / 256)), thr, 0, stream>>>(s, d, n);
  };
  cvt(sa_wq, wq_s, WH); cvt(sa_wk, wk_s, WH); cvt(sa_wv, wv_s, WH);
  cvt(sa_wo, wo_s, WH); cvt(ca_wq, wq_c, WH); cvt(ca_wk, wk_c, WH);
  cvt(ca_wv, wv_c, WH); cvt(ca_wo, wo_c, WH);
  cvt(ff_w1, w1b, WH * 4); cvt(ff_w2, w2b, WH * 4);
  cvt(x, abfA, kACT);
  cvt(enc, encb, kACT);

  auto gemm_bf = [&](const u16* A, const u16* W, const float* bias, u16* C,
                     int Nc, int K) {
    gemm_bf16_kernel<0><<<dim3(Nc / 128, Mr / 128), thr, 0, stream>>>(
        A, W, bias, nullptr, nullptr, C, Mr, Nc, K);
  };
  auto gemm_br = [&](const u16* A, const u16* W, const float* bias,
                     const float* res, float* C, int Nc, int K) {
    gemm_bf16_kernel<1><<<dim3(Nc / 128, Mr / 128), thr, 0, stream>>>(
        A, W, bias, res, C, nullptr, Mr, Nc, K);
  };
  auto gemm_gelu = [&](const u16* A, const u16* W, const float* bias, u16* C,
                       int Nc, int K) {
    gemm_bf16_kernel<2><<<dim3(Nc / 128, Mr / 128), thr, 0, stream>>>(
        A, W, bias, nullptr, nullptr, C, Mr, Nc, K);
  };

  auto attention = [&](const u16* qin, const u16* kvin, const u16* wqb,
                       const float* bq, const u16* wkb, const float* bk,
                       const u16* wvb, const float* bv, const float* proj) {
    gemm_bf(qin, wqb, bq, qb, kH, kH);
    gemm_bf(kvin, wkb, bk, kb, kH, kH);
    gemm_bf(kvin, wvb, bv, vb, kH, kH);
    feat_kernel<0><<<dim3(kBH, kN / 64), thr, 0, stream>>>(qb, proj, Qp);
    feat_kernel<1><<<dim3(kBH, kN / 64), thr, 0, stream>>>(kb, proj, KpT);
    ksum_kernel<<<dim3(kBH * kM), thr, 0, stream>>>(KpT, Ksum);
    ctx_kernel<<<dim3(kBH, kKSPLIT), thr, 0, stream>>>(KpT, vb, ctxp);
    ctx_reduce_kernel<<<dim3((unsigned)(kCTX / 256)), thr, 0, stream>>>(ctxp, ctxb);
    attn_out_kernel<<<dim3(kBH, kN / 128), thr, 0, stream>>>(Qp, ctxb, Ksum, abfB);
  };

  // --- self attention + LN1 ---
  attention(abfA, abfA, wq_s, sa_bq, wk_s, sa_bk, wv_s, sa_bv, sa_proj);
  gemm_br(abfB, wo_s, sa_bo, x, t32, kH, kH);
  ln_kernel<<<dim3(Mr), thr, 0, stream>>>(t32, ln1_g, ln1_b, h1, abfA);

  // --- cross attention + LN2 ---
  attention(abfA, encb, wq_c, ca_bq, wk_c, ca_bk, wv_c, ca_bv, ca_proj);
  gemm_br(abfB, wo_c, ca_bo, h1, t32, kH, kH);
  ln_kernel<<<dim3(Mr), thr, 0, stream>>>(t32, ln2_g, ln2_b, h2, abfA);

  // --- feed-forward + LN3 ---
  gemm_gelu(abfA, w1b, ff_b1, gelub, 4 * kH, kH);
  gemm_br(gelub, w2b, ff_b2, h2, t32, kH, 4 * kH);
  ln_kernel<<<dim3(Mr), thr, 0, stream>>>(t32, ln3_g, ln3_b, out, nullptr);

  (void)in_sizes; (void)n_in; (void)out_size; (void)ws_size;
}
